// FeedForwardQuantum_65481071408436
// MI455X (gfx1250) — matrix operand order repaired and checked by reading, not compiled, not run on hardware
//
#include <hip/hip_runtime.h>

typedef unsigned short ushort_t;
typedef unsigned int   uint_t;

typedef __attribute__((ext_vector_type(16))) __bf16 v16bf;
typedef __attribute__((ext_vector_type(8)))  float  v8f;

#define EMB   1024
#define FFND  4096
#define NQ    10
#define NROWS 16384   // B*S = 8*2048

#if defined(__has_builtin)
#  if __has_builtin(__builtin_amdgcn_global_load_async_to_lds_b128)
#    define HAVE_ASYNC_LDS 1
#  endif
#  if __has_builtin(__builtin_amdgcn_s_wait_asynccnt)
#    define HAVE_WAIT_ASYNC 1
#  endif
#endif

// typed pointers matching the builtin's signature:
//   (v4i addrspace(1)* global_src, v4i addrspace(3)* lds_dst, imm offset, imm cpol)
typedef int v4i __attribute__((vector_size(16)));
typedef v4i __attribute__((address_space(1))) g_v4i;
typedef v4i __attribute__((address_space(3))) l_v4i;
typedef __attribute__((address_space(3))) void lds_void_t;

__device__ __forceinline__ void async_ld_b128(const ushort_t* g, ushort_t* l) {
#if defined(HAVE_ASYNC_LDS)
    __builtin_amdgcn_global_load_async_to_lds_b128((g_v4i*)g, (l_v4i*)l, 0, 0);
#else
    unsigned loff = (unsigned)(size_t)(lds_void_t*)l;   // LDS byte offset
    asm volatile("global_load_async_to_lds_b128 %0, %1, off"
                 :: "v"(loff), "v"((unsigned long long)g)
                 : "memory");
#endif
}

__device__ __forceinline__ void wait_async8() {
#if defined(HAVE_WAIT_ASYNC)
    __builtin_amdgcn_s_wait_asynccnt(8);
#else
    asm volatile("s_wait_asynccnt 8" ::: "memory");
#endif
}
__device__ __forceinline__ void wait_async0() {
#if defined(HAVE_WAIT_ASYNC)
    __builtin_amdgcn_s_wait_asynccnt(0);
#else
    asm volatile("s_wait_asynccnt 0" ::: "memory");
#endif
}

// ---------------- bf16 helpers ----------------
__device__ __forceinline__ ushort_t f2bf(float f) {
    uint_t u = __float_as_uint(f);
    u += 0x7FFFu + ((u >> 16) & 1u);   // round-to-nearest-even
    return (ushort_t)(u >> 16);
}

// ---------------- kernel 1: W2 fp32 -> bf16 ----------------
__global__ void w2_to_bf16(const float* __restrict__ W2, ushort_t* __restrict__ W2b) {
    int i = (blockIdx.x * 256 + threadIdx.x) * 4;
    float4 v = *(const float4*)(W2 + i);
    unsigned long long pack =
        (unsigned long long)f2bf(v.x)        |
        ((unsigned long long)f2bf(v.y) << 16) |
        ((unsigned long long)f2bf(v.z) << 32) |
        ((unsigned long long)f2bf(v.w) << 48);
    *(unsigned long long*)(W2b + i) = pack;
}

// ---------------- kernel 2: fused front (proj -> cos*cos -> FFN1 -> relu -> bf16 h) ----------------
__global__ void front_kernel(const float* __restrict__ x,
                             const float* __restrict__ Win,
                             const float* __restrict__ b_in,
                             const float* __restrict__ theta,
                             const float* __restrict__ W1,
                             const float* __restrict__ b1,
                             ushort_t* __restrict__ hout) {
    const int wave = threadIdx.x >> 5;
    const int lane = threadIdx.x & 31;
    const int row  = blockIdx.x * 8 + wave;   // one wave32 per row

    const float* xr = x + (size_t)row * EMB;

    float part[NQ];
#pragma unroll
    for (int q = 0; q < NQ; ++q) part[q] = 0.f;

    for (int j = 0; j < EMB / 32; ++j) {
        float xv = xr[j * 32 + lane];
#pragma unroll
        for (int q = 0; q < NQ; ++q)
            part[q] += xv * Win[q * EMB + j * 32 + lane];
    }
    // full-wave butterfly reduction (wave32)
#pragma unroll
    for (int off = 16; off > 0; off >>= 1) {
#pragma unroll
        for (int q = 0; q < NQ; ++q)
            part[q] += __shfl_xor(part[q], off, 32);
    }

    float qc[NQ];
#pragma unroll
    for (int q = 0; q < NQ; ++q)
        qc[q] = cosf(part[q] + b_in[q]) * cosf(theta[q]);

    ushort_t* hr = hout + (size_t)row * FFND;
    for (int i = 0; i < FFND / 32; ++i) {
        int f = i * 32 + lane;
        float acc = b1[f];
#pragma unroll
        for (int q = 0; q < NQ; ++q)
            acc += qc[q] * W1[f * NQ + q];
        acc = acc > 0.f ? acc : 0.f;     // relu
        hr[f] = f2bf(acc);
    }
}

// ---------------- kernel 3: out = h(bf16) @ W2bf^T + b2, WMMA bf16, async double-buffer ----------------
#define BM  128
#define BN  128
#define BK  64
#define LDT 72   // padded LDS row length in bf16 elems

union Frag { v16bf v; uint_t u[8]; };

__global__ __launch_bounds__(256) void gemm_kernel(const ushort_t* __restrict__ A,
                                                   const ushort_t* __restrict__ B,
                                                   const float* __restrict__ b2,
                                                   float* __restrict__ out) {
    __shared__ ushort_t As[2][BM * LDT];   // 2 x 18 KB
    __shared__ ushort_t Bs[2][BN * LDT];   // 2 x 18 KB

    const int tid  = threadIdx.x;
    const int wave = tid >> 5;
    const int lane = tid & 31;
    const int hl   = lane >> 4;     // half of wave
    const int l16  = lane & 15;

    const int m0 = blockIdx.y * BM;
    const int n0 = blockIdx.x * BN;
    const int wm = (wave & 3) * 32;  // 4 waves tile M
    const int wn = (wave >> 2) * 64; // 2 waves tile N

    v8f zero = {0.f, 0.f, 0.f, 0.f, 0.f, 0.f, 0.f, 0.f};
    v8f acc[2][4];
#pragma unroll
    for (int i = 0; i < 2; ++i)
#pragma unroll
        for (int j = 0; j < 4; ++j) acc[i][j] = zero;

    // cooperative tile fill indices: 256 thr * 16B = 4KB/pass, 4 passes per 16KB tile
    const int lr = tid >> 3;         // 0..31 (row within pass)
    const int lk = (tid & 7) * 8;    // bf16 col offset, 16B granules

    // fragment K offsets per ISA 16-bit layouts
    int akoff[8], bkoff[8];
#pragma unroll
    for (int v = 0; v < 8; ++v) {
        akoff[v] = 2 * v + ((v >= 4) ? 8 : 0) + (hl ? 8 : 0);  // A: 16x32 bf16
        bkoff[v] = 2 * v + ((v >= 4) ? 8 : 0) + (hl ? 8 : 0);  // B: 32x16 bf16 (K-major rows)
    }

    const ushort_t* Ag = A + (size_t)m0 * FFND;
    const ushort_t* Bg = B + (size_t)n0 * FFND;

    // 8 async b128 fills per thread per tile (4 for A, 4 for B)
    auto fill = [&](int buf, int k0) {
#pragma unroll
        for (int p = 0; p < 4; ++p) {
            int r = lr + 32 * p;
            async_ld_b128(&Ag[(size_t)r * FFND + k0 + lk], &As[buf][r * LDT + lk]);
            async_ld_b128(&Bg[(size_t)r * FFND + k0 + lk], &Bs[buf][r * LDT + lk]);
        }
    };

    const int nsteps = FFND / BK;    // 64
    fill(0, 0);

    for (int s = 0; s < nsteps; ++s) {
        const int buf = s & 1;
        if (s + 1 < nsteps) {
            fill(buf ^ 1, (s + 1) * BK);   // prefetch next tile into other buffer
            wait_async8();                 // in-order: my 8 fills for tile s are done
        } else {
            wait_async0();
        }
        __syncthreads();                   // whole tile s visible to all waves

#pragma unroll
        for (int kk = 0; kk < BK; kk += 32) {
            Frag af[2], bfr[4];
#pragma unroll
            for (int i = 0; i < 2; ++i)
#pragma unroll
                for (int v = 0; v < 8; ++v)
                    af[i].u[v] = *(const uint_t*)&As[buf][(wm + i * 16 + l16) * LDT + kk + akoff[v]];
#pragma unroll
            for (int j = 0; j < 4; ++j)
#pragma unroll
                for (int v = 0; v < 8; ++v)
                    bfr[j].u[v] = *(const uint_t*)&Bs[buf][(wn + j * 16 + l16) * LDT + kk + bkoff[v]];
#pragma unroll
            for (int i = 0; i < 2; ++i)
#pragma unroll
                for (int j = 0; j < 4; ++j)
                    acc[i][j] = __builtin_amdgcn_wmma_f32_16x16x32_bf16(
                        false, af[i].v, false, bfr[j].v,
                        (short)0, acc[i][j], false, false);
        }
        __syncthreads();                   // all reads of buffer done before it is refilled
    }

    // epilogue: C layout — vgpr r, lanes 0-15: M=r, lanes 16-31: M=r+8; N = l16
#pragma unroll
    for (int i = 0; i < 2; ++i) {
#pragma unroll
        for (int j = 0; j < 4; ++j) {
            int Nc = n0 + wn + j * 16 + l16;
            float bias = b2[Nc];
#pragma unroll
            for (int r = 0; r < 8; ++r) {
                int Mr = m0 + wm + i * 16 + r + (hl ? 8 : 0);
                out[(size_t)Mr * EMB + Nc] = acc[i][j][r] + bias;
            }
        }
    }
}

// ---------------- launch ----------------
extern "C" void kernel_launch(void* const* d_in, const int* in_sizes, int n_in,
                              void* d_out, int out_size, void* d_ws, size_t ws_size,
                              hipStream_t stream) {
    const float* x     = (const float*)d_in[0];
    const float* Win   = (const float*)d_in[1];
    const float* b_in  = (const float*)d_in[2];
    const float* theta = (const float*)d_in[3];
    const float* W1    = (const float*)d_in[4];
    const float* b1    = (const float*)d_in[5];
    const float* W2    = (const float*)d_in[6];
    const float* b2    = (const float*)d_in[7];
    float* out = (float*)d_out;

    ushort_t* hbuf = (ushort_t*)d_ws;                        // 16384*4096*2 = 128 MB
    ushort_t* w2b  = hbuf + (size_t)NROWS * FFND;            // + 8 MB

    hipLaunchKernelGGL(w2_to_bf16, dim3((EMB * FFND) / (256 * 4)), dim3(256), 0, stream,
                       W2, w2b);
    hipLaunchKernelGGL(front_kernel, dim3(NROWS / 8), dim3(256), 0, stream,
                       x, Win, b_in, theta, W1, b1, hbuf);
    hipLaunchKernelGGL(gemm_kernel, dim3(EMB / BN, NROWS / BM), dim3(256), 0, stream,
                       hbuf, w2b, b2, out);
}
